// Model_56856777064948
// MI455X (gfx1250) — compile-verified, matrix operand order checked by reading, not run on hardware
//
#include <hip/hip_runtime.h>

// ---------------------------------------------------------------------------
// Types / helpers
// ---------------------------------------------------------------------------
typedef __bf16 bf16;
typedef __attribute__((ext_vector_type(16))) __bf16 v16bf;
typedef __attribute__((ext_vector_type(8)))  __bf16 v8bf;
typedef __attribute__((ext_vector_type(8)))  float  v8f;
typedef __attribute__((ext_vector_type(4)))  float  v4f;

#define DEV static __device__ __forceinline__

static constexpr int P256 = 264;   // padded LDS row stride for 256-wide tiles
static constexpr int P128 = 136;   // padded LDS row stride for 128-wide tiles
static constexpr int PGX  = 328;   // padded LDS row stride for K<=320 staging

DEV bf16 f2bf(float f) {
    union { float f; unsigned u; } a; a.f = f;
    unsigned u = a.u;
    unsigned r = u + 0x7fffu + ((u >> 16) & 1u);
    union { unsigned short s; bf16 b; } o; o.s = (unsigned short)(r >> 16);
    return o.b;
}
DEV float bf2f(bf16 b) {
    union { unsigned short s; bf16 b; } i; i.b = b;
    union { unsigned u; float f; } o; o.u = ((unsigned)i.s) << 16;
    return o.f;
}
DEV float sigm(float x) { return 1.f / (1.f + __expf(-x)); }

// Optimization barrier: returns an opaque zero. Adding it to a kernel-arg
// base pointer keeps global-addrspace inference intact (loads stay
// global_load_b128, NOT flat_load) while preventing LICM from hoisting the
// weight loads out of the time loop (which caused scratch spills earlier).
DEV size_t lz() {
    size_t v = 0;
    asm volatile("" : "+s"(v));
    return v;
}

// 16-bit WMMA operand fragment (A or B): lane holds row (m or n),
// 16 bf16 at K = {k0+8g .. +7} and {k0+16+8g .. +7}.
DEV v16bf ldfrag(const bf16* row, int k0, int g) {
    union { v16bf v; v8bf h[2]; } u;
    u.h[0] = *(const v8bf*)(row + k0 + 8 * g);
    u.h[1] = *(const v8bf*)(row + k0 + 16 + 8 * g);
    return u.v;
}

DEV v8f wmma_bf(v16bf a, v16bf b, v8f c) {
    return __builtin_amdgcn_wmma_f32_16x16x32_bf16(false, a, false, b,
                                                   (short)0, c, false, false);
}

// ---------------------------------------------------------------------------
// Small utility kernels
// ---------------------------------------------------------------------------
__global__ void k_cast(const float* __restrict__ src, bf16* __restrict__ dst, int n) {
    int i = blockIdx.x * 256 + threadIdx.x;
    if (i < n) dst[i] = f2bf(src[i]);
}
__global__ void k_zero_u8(unsigned char* p, int n) {
    int i = blockIdx.x * 256 + threadIdx.x;
    if (i < n) p[i] = 0;
}
__global__ void k_copy_f32(const float* __restrict__ s, float* __restrict__ d, int n) {
    int i = blockIdx.x * 256 + threadIdx.x;
    if (i < n) d[i] = s[i];
}

// ---------------------------------------------------------------------------
// Persistent imputation RNN. One WG = 16 batch rows, 8 waves, T loop inside.
// ---------------------------------------------------------------------------
__global__ void __launch_bounds__(256)
k_impute(const float* __restrict__ xdyn, const float* __restrict__ xmask,
         const float* __restrict__ xdelta,
         const bf16* __restrict__ gammaW, const float* __restrict__ gamma_b,
         const bf16* __restrict__ ilinW,  const float* __restrict__ ilin_b,
         const bf16* __restrict__ Wih,    const bf16* __restrict__ Whh,
         const float* __restrict__ bih,   const float* __restrict__ bhh,
         float* __restrict__ out_xh, float* __restrict__ out_xc,
         bf16* __restrict__ xc_bf /*[T,B,128]*/)
{
    __shared__ __align__(16) bf16  sh_h[16 * P256];
    __shared__ __align__(16) float sh_c[16 * P256];
    __shared__ __align__(16) float sh_x[16 * P128];    // x, later reused for xh
    __shared__ __align__(16) float sh_m[16 * P128];    // mask, later reused for xc
    __shared__ __align__(16) bf16  sh_d[16 * P128];
    __shared__ __align__(16) bf16  sh_xb[16 * P128];
    __shared__ __align__(16) bf16  sh_xcm[16 * P256];  // [xc | mask] bf16

    const int tid  = threadIdx.x;
    const int lane = tid & 31, wave = tid >> 5;
    const int g = lane >> 4, ln = lane & 15;
    const int m0 = blockIdx.x * 16;

    for (int i = tid; i < 16 * P256; i += 256) { sh_h[i] = f2bf(0.f); sh_c[i] = 0.f; }
    __syncthreads();

    for (int t = 0; t < 128; ++t) {
        const size_t z = lz();                 // per-iteration opaque zero
        const bf16* gW_t  = gammaW + z;
        const bf16* ilW_t = ilinW  + z;
        const bf16* Wih_t = Wih    + z;
        const bf16* Whh_t = Whh    + z;

        // ---- Phase A: stage x, mask, delta ----
        for (int i = tid; i < 16 * 128; i += 256) {
            int m = i >> 7, j = i & 127;
            size_t gi = ((size_t)(m0 + m) * 128 + t) * 128 + j;   // [B,T,D]
            float xv = xdyn[gi], mv = xmask[gi], dv = xdelta[gi];
            sh_x[m * P128 + j] = xv; sh_m[m * P128 + j] = mv;
            sh_xb[m * P128 + j] = f2bf(xv); sh_d[m * P128 + j] = f2bf(dv);
            sh_xcm[m * P256 + 128 + j] = f2bf(mv);
        }
        __syncthreads();

        // ---- Phase B: gamma = exp(-relu(d @ gW^T + b)); h *= gamma ----
        {
            int n0 = wave * 32;
            #pragma unroll
            for (int s = 0; s < 2; ++s) {
                v8f acc = {};
                const bf16* wr = gW_t + (size_t)(n0 + s * 16 + ln) * 128;
                #pragma unroll
                for (int ks = 0; ks < 4; ++ks)
                    acc = wmma_bf(ldfrag(sh_d + ln * P128, ks * 32, g),
                                  ldfrag(wr, ks * 32, g), acc);
                int n = n0 + s * 16 + ln;
                float bv = gamma_b[n];
                #pragma unroll
                for (int e = 0; e < 8; ++e) {
                    int m = e + 8 * g;
                    float gam = __expf(-fmaxf(acc[e] + bv, 0.f));
                    int idx = m * P256 + n;
                    sh_h[idx] = f2bf(bf2f(sh_h[idx]) * gam);
                }
            }
        }
        __syncthreads();

        // ---- Phase C: xh = [x,h] @ ilinW^T + b ; xc = m*x+(1-m)*xh ----
        {
            v8f acc = {};
            int n0 = wave * 16;
            const bf16* wr = ilW_t + (size_t)(n0 + ln) * 384;
            #pragma unroll
            for (int ks = 0; ks < 4; ++ks)
                acc = wmma_bf(ldfrag(sh_xb + ln * P128, ks * 32, g),
                              ldfrag(wr, ks * 32, g), acc);
            #pragma unroll
            for (int ks = 0; ks < 8; ++ks)
                acc = wmma_bf(ldfrag(sh_h + ln * P256, ks * 32, g),
                              ldfrag(wr, 128 + ks * 32, g), acc);
            int n = n0 + ln;
            float bv = ilin_b[n];
            #pragma unroll
            for (int e = 0; e < 8; ++e) {
                int m = e + 8 * g;
                float xh = acc[e] + bv;
                float xv = sh_x[m * P128 + n], mv = sh_m[m * P128 + n];
                float xc = mv * xv + (1.f - mv) * xh;
                sh_x[m * P128 + n] = xh;          // reuse as staging for out_xh
                sh_m[m * P128 + n] = xc;          // reuse as staging for out_xc
                sh_xcm[m * P256 + n] = f2bf(xc);
            }
        }
        __syncthreads();

        // ---- coalesced stores of xh / xc (f32) and xc (bf16) ----
        for (int i = tid * 4; i < 16 * 128; i += 256 * 4) {
            int m = i >> 7, j = i & 127;
            size_t go = ((size_t)(m0 + m) * 128 + t) * 128 + j;
            *(v4f*)(out_xh + go) = *(const v4f*)(sh_x + m * P128 + j);
            *(v4f*)(out_xc + go) = *(const v4f*)(sh_m + m * P128 + j);
        }
        {
            int i = tid * 8;           // 2048 bf16 total, exactly one chunk/thread
            int m = i >> 7, j = i & 127;
            *(v8bf*)(xc_bf + ((size_t)t * 256 + m0 + m) * 128 + j) =
                *(const v8bf*)(sh_xcm + m * P256 + j);
        }

        // ---- Phase D: gates = [xc,m]@Wih^T + h@Whh^T; LSTM cell ----
        {
            v8f acc[8];
            int nc0 = wave * 32;
            #pragma unroll
            for (int gi = 0; gi < 4; ++gi)
                #pragma unroll
                for (int s = 0; s < 2; ++s) {
                    int row = gi * 256 + nc0 + s * 16 + ln;
                    const bf16* wih = Wih_t + (size_t)row * 256;
                    const bf16* whh = Whh_t + (size_t)row * 256;
                    v8f a = {};
                    #pragma unroll
                    for (int ks = 0; ks < 8; ++ks)
                        a = wmma_bf(ldfrag(sh_xcm + ln * P256, ks * 32, g),
                                    ldfrag(wih, ks * 32, g), a);
                    #pragma unroll
                    for (int ks = 0; ks < 8; ++ks)
                        a = wmma_bf(ldfrag(sh_h + ln * P256, ks * 32, g),
                                    ldfrag(whh, ks * 32, g), a);
                    acc[gi * 2 + s] = a;
                }
            __syncthreads();   // all reads of sh_h / sh_xcm done before h update
            #pragma unroll
            for (int s = 0; s < 2; ++s) {
                int n = nc0 + s * 16 + ln;
                float bi  = bih[n]       + bhh[n];
                float bf_ = bih[256 + n] + bhh[256 + n];
                float bg  = bih[512 + n] + bhh[512 + n];
                float bo  = bih[768 + n] + bhh[768 + n];
                #pragma unroll
                for (int e = 0; e < 8; ++e) {
                    int m = e + 8 * g;
                    float iv = sigm(acc[0 * 2 + s][e] + bi);
                    float fv = sigm(acc[1 * 2 + s][e] + bf_);
                    float gv = tanhf(acc[2 * 2 + s][e] + bg);
                    float ov = sigm(acc[3 * 2 + s][e] + bo);
                    int idx = m * P256 + n;
                    float cn = fv * sh_c[idx] + iv * gv;
                    sh_c[idx] = cn;
                    sh_h[idx] = f2bf(ov * tanhf(cn));
                }
            }
        }
        __syncthreads();
    }
}

// ---------------------------------------------------------------------------
// Persistent LSTM layer (templated on input width for full unroll).
// ---------------------------------------------------------------------------
template <int DIN>
__global__ void __launch_bounds__(256)
k_lstm_t(const bf16* __restrict__ xin /*[T,B,DIN]*/,
         const bf16* __restrict__ Wih, const bf16* __restrict__ Whh,
         const float* __restrict__ bih, const float* __restrict__ bhh,
         bf16* __restrict__ out /*[T,B,256]*/)
{
    __shared__ __align__(16) bf16  sh_h[16 * P256];
    __shared__ __align__(16) float sh_c[16 * P256];
    __shared__ __align__(16) bf16  sh_xin[16 * P256];

    const int tid = threadIdx.x;
    const int lane = tid & 31, wave = tid >> 5;
    const int g = lane >> 4, ln = lane & 15;
    const int m0 = blockIdx.x * 16;

    for (int i = tid; i < 16 * P256; i += 256) { sh_h[i] = f2bf(0.f); sh_c[i] = 0.f; }
    __syncthreads();

    for (int t = 0; t < 128; ++t) {
        const size_t z = lz();
        const bf16* Wih_t = Wih + z;
        const bf16* Whh_t = Whh + z;

        // stage input tile (coalesced b128)
        for (int i = tid * 8; i < 16 * DIN; i += 256 * 8) {
            int m = i / DIN, j = i - m * DIN;
            *(v8bf*)(sh_xin + m * P256 + j) =
                *(const v8bf*)(xin + ((size_t)t * 256 + m0 + m) * DIN + j);
        }
        __syncthreads();

        v8f acc[8];
        int nc0 = wave * 32;
        #pragma unroll
        for (int gi = 0; gi < 4; ++gi)
            #pragma unroll
            for (int s = 0; s < 2; ++s) {
                int row = gi * 256 + nc0 + s * 16 + ln;
                const bf16* wih = Wih_t + (size_t)row * DIN;
                const bf16* whh = Whh_t + (size_t)row * 256;
                v8f a = {};
                #pragma unroll
                for (int ks = 0; ks < DIN / 32; ++ks)
                    a = wmma_bf(ldfrag(sh_xin + ln * P256, ks * 32, g),
                                ldfrag(wih, ks * 32, g), a);
                #pragma unroll
                for (int ks = 0; ks < 8; ++ks)
                    a = wmma_bf(ldfrag(sh_h + ln * P256, ks * 32, g),
                                ldfrag(whh, ks * 32, g), a);
                acc[gi * 2 + s] = a;
            }
        __syncthreads();
        #pragma unroll
        for (int s = 0; s < 2; ++s) {
            int n = nc0 + s * 16 + ln;
            float bi  = bih[n]       + bhh[n];
            float bf_ = bih[256 + n] + bhh[256 + n];
            float bg  = bih[512 + n] + bhh[512 + n];
            float bo  = bih[768 + n] + bhh[768 + n];
            #pragma unroll
            for (int e = 0; e < 8; ++e) {
                int m = e + 8 * g;
                float iv = sigm(acc[0 * 2 + s][e] + bi);
                float fv = sigm(acc[1 * 2 + s][e] + bf_);
                float gv = tanhf(acc[2 * 2 + s][e] + bg);
                float ov = sigm(acc[3 * 2 + s][e] + bo);
                int idx = m * P256 + n;
                float cn = fv * sh_c[idx] + iv * gv;
                sh_c[idx] = cn;
                sh_h[idx] = f2bf(ov * tanhf(cn));
            }
        }
        __syncthreads();
        // coalesced output store of h tile
        for (int i = tid * 8; i < 16 * 256; i += 256 * 8) {
            int m = i >> 8, j = i & 255;
            *(v8bf*)(out + ((size_t)t * 256 + m0 + m) * 256 + j) =
                *(const v8bf*)(sh_h + m * P256 + j);
        }
    }
}

// ---------------------------------------------------------------------------
// Feature build: feat[b*T+t] = [rnn_out(t,b,:256) | static(b,:64)]  (bf16)
// ---------------------------------------------------------------------------
__global__ void k_feat(const bf16* __restrict__ l2out, const float* __restrict__ xstatic,
                       bf16* __restrict__ feat)
{
    int idx = blockIdx.x * 256 + threadIdx.x;
    if (idx >= 32768 * 320) return;
    int r = idx / 320, j = idx - r * 320;
    int b = r >> 7, t = r & 127;
    feat[idx] = (j < 256) ? l2out[((size_t)t * 256 + b) * 256 + j]
                          : f2bf(xstatic[b * 64 + (j - 256)]);
}

// ---------------------------------------------------------------------------
// WMMA GEMM: Y = X[R,K] @ W[N,K]^T + bias. Block = 16 rows, 8 waves cover N.
// mode 0: relu -> bf16 (N==256) ; mode 1: raw fp32 (heads, small N)
// ---------------------------------------------------------------------------
__global__ void __launch_bounds__(256)
k_gemm(const bf16* __restrict__ X, int K, const bf16* __restrict__ W,
       const float* __restrict__ bias, int N,
       bf16* __restrict__ Ybf, float* __restrict__ Yf32, int mode)
{
    __shared__ __align__(16) bf16 sh_x[16 * PGX];
    __shared__ __align__(16) bf16 sh_o[16 * P256];
    const int tid = threadIdx.x, lane = tid & 31, wave = tid >> 5;
    const int g = lane >> 4, ln = lane & 15;
    const int r0 = blockIdx.x * 16;

    for (int i = tid * 8; i < 16 * K; i += 256 * 8) {
        int m = i / K, j = i - m * K;
        *(v8bf*)(sh_x + m * PGX + j) = *(const v8bf*)(X + (size_t)(r0 + m) * K + j);
    }
    __syncthreads();

    const int CT = N >> 4;
    const int CPW = (CT + 7) >> 3;
    for (int q = 0; q < CPW; ++q) {
        int ct = wave * CPW + q;
        if (ct >= CT) break;             // wave-uniform exit
        v8f acc = {};
        const bf16* wr = W + (size_t)(ct * 16 + ln) * K;
        for (int ks = 0; ks < (K >> 5); ++ks)
            acc = wmma_bf(ldfrag(sh_x + ln * PGX, ks * 32, g),
                          ldfrag(wr, ks * 32, g), acc);
        int n = ct * 16 + ln;
        float bv = bias[n];
        if (mode == 0) {
            #pragma unroll
            for (int e = 0; e < 8; ++e)
                sh_o[(e + 8 * g) * P256 + n] = f2bf(fmaxf(acc[e] + bv, 0.f));
        } else {
            #pragma unroll
            for (int e = 0; e < 8; ++e)
                Yf32[(size_t)(r0 + e + 8 * g) * N + n] = acc[e] + bv;
        }
    }
    if (mode == 0) {
        __syncthreads();
        for (int i = tid * 8; i < 16 * 256; i += 256 * 8) {
            int m = i >> 8, j = i & 255;
            *(v8bf*)(Ybf + (size_t)(r0 + m) * N + j) = *(const v8bf*)(sh_o + m * P256 + j);
        }
    }
}

// ---------------------------------------------------------------------------
// Finalize: sigmoid(aux), relu(att), exp_, y_proba
// ---------------------------------------------------------------------------
__global__ void k_finalize(const float* __restrict__ aux_raw, const float* __restrict__ att_raw,
                           const float* __restrict__ xstatic, const float* __restrict__ xdyn,
                           float* __restrict__ o_aux, float* __restrict__ o_att,
                           float* __restrict__ o_yp, float* __restrict__ o_exp)
{
    int r = blockIdx.x * 256 + threadIdx.x;
    if (r >= 32768) return;
    int b = r >> 7, t = r & 127;
    float e[40];
    for (int i = 0; i < 16; ++i) {
        float p = sigm(aux_raw[(size_t)r * 16 + i]);
        o_aux[(size_t)r * 16 + i] = p;
        e[i] = p;
    }
    for (int i = 0; i < 8; ++i)  e[16 + i] = xstatic[b * 64 + i];
    for (int i = 0; i < 16; ++i) e[24 + i] = xdyn[((size_t)b * 128 + t) * 128 + i];
    float s = 0.f;
    for (int i = 0; i < 40; ++i) {
        float av = fmaxf(att_raw[(size_t)r * 48 + i], 0.f);
        o_att[(size_t)r * 40 + i] = av;
        o_exp[(size_t)r * 40 + i] = e[i];
        s += e[i] * av;
    }
    o_yp[r] = 1.f - __expf(-s);
}

// ---------------------------------------------------------------------------
// Host orchestration
// ---------------------------------------------------------------------------
extern "C" void kernel_launch(void* const* d_in, const int* in_sizes, int n_in,
                              void* d_out, int out_size, void* d_ws, size_t ws_size,
                              hipStream_t stream)
{
    (void)in_sizes; (void)n_in; (void)out_size; (void)ws_size;
    const float* xdyn    = (const float*)d_in[0];
    const float* xmask   = (const float*)d_in[1];
    const float* xdelta  = (const float*)d_in[2];
    const float* xstatic = (const float*)d_in[3];

    char* ws = (char*)d_ws;
    size_t off = 0;
    auto alloc = [&](size_t bytes) -> void* {
        void* p = ws + off;
        off = (off + bytes + 255) & ~(size_t)255;
        return p;
    };
    bf16* gW    = (bf16*)alloc((size_t)256 * 128 * 2);
    bf16* ilW   = (bf16*)alloc((size_t)128 * 384 * 2);
    bf16* iWih  = (bf16*)alloc((size_t)1024 * 256 * 2);
    bf16* iWhh  = (bf16*)alloc((size_t)1024 * 256 * 2);
    bf16* lWih[3]; bf16* lWhh[3];
    lWih[0] = (bf16*)alloc((size_t)1024 * 128 * 2);
    lWhh[0] = (bf16*)alloc((size_t)1024 * 256 * 2);
    lWih[1] = (bf16*)alloc((size_t)1024 * 256 * 2);
    lWhh[1] = (bf16*)alloc((size_t)1024 * 256 * 2);
    lWih[2] = (bf16*)alloc((size_t)1024 * 256 * 2);
    lWhh[2] = (bf16*)alloc((size_t)1024 * 256 * 2);
    bf16* aux0 = (bf16*)alloc((size_t)256 * 320 * 2);
    bf16* aux1 = (bf16*)alloc((size_t)256 * 256 * 2);
    bf16* aux2 = (bf16*)alloc((size_t)256 * 256 * 2);
    bf16* auxO = (bf16*)alloc((size_t)16 * 256 * 2);
    bf16* att0 = (bf16*)alloc((size_t)256 * 320 * 2);
    bf16* att1 = (bf16*)alloc((size_t)256 * 256 * 2);
    bf16* att2 = (bf16*)alloc((size_t)256 * 256 * 2);
    bf16* attO = (bf16*)alloc((size_t)48 * 256 * 2);    // padded 40->48 rows
    float* attB = (float*)alloc((size_t)48 * 4);        // padded bias
    bf16* xc_bf = (bf16*)alloc((size_t)128 * 256 * 128 * 2);  // [T,B,128]
    bf16* lbuf0 = (bf16*)alloc((size_t)128 * 256 * 256 * 2);  // [T,B,256]
    bf16* lbuf1 = (bf16*)alloc((size_t)128 * 256 * 256 * 2);
    bf16* feat  = (bf16*)alloc((size_t)32768 * 320 * 2);
    bf16* mp0   = (bf16*)alloc((size_t)32768 * 256 * 2);
    bf16* mp1   = (bf16*)alloc((size_t)32768 * 256 * 2);
    float* aux_raw = (float*)alloc((size_t)32768 * 16 * 4);
    float* att_raw = (float*)alloc((size_t)32768 * 48 * 4);

    auto cast = [&](const void* src, bf16* dst, int n) {
        k_cast<<<dim3((n + 255) / 256), dim3(256), 0, stream>>>((const float*)src, dst, n);
    };

    k_zero_u8<<<dim3((48 * 256 * 2 + 255) / 256), dim3(256), 0, stream>>>((unsigned char*)attO, 48 * 256 * 2);
    k_zero_u8<<<dim3(1), dim3(256), 0, stream>>>((unsigned char*)attB, 48 * 4);

    cast(d_in[4],  gW,    256 * 128);
    cast(d_in[6],  ilW,   128 * 384);
    cast(d_in[8],  iWih,  1024 * 256);
    cast(d_in[9],  iWhh,  1024 * 256);
    cast(d_in[12], lWih[0], 1024 * 128);
    cast(d_in[13], lWhh[0], 1024 * 256);
    cast(d_in[16], lWih[1], 1024 * 256);
    cast(d_in[17], lWhh[1], 1024 * 256);
    cast(d_in[20], lWih[2], 1024 * 256);
    cast(d_in[21], lWhh[2], 1024 * 256);
    cast(d_in[24], aux0, 256 * 320);
    cast(d_in[26], aux1, 256 * 256);
    cast(d_in[28], aux2, 256 * 256);
    cast(d_in[30], auxO, 16 * 256);
    cast(d_in[32], att0, 256 * 320);
    cast(d_in[34], att1, 256 * 256);
    cast(d_in[36], att2, 256 * 256);
    cast(d_in[38], attO, 40 * 256);   // first 40 rows contiguous in padded buffer
    k_copy_f32<<<dim3(1), dim3(256), 0, stream>>>((const float*)d_in[39], attB, 40);

    float* out   = (float*)d_out;
    float* o_xh  = out;
    float* o_xc  = out + (size_t)4194304;
    float* o_aux = out + (size_t)8388608;
    float* o_att = out + (size_t)8912896;
    float* o_yp  = out + (size_t)10223616;
    float* o_exp = out + (size_t)10256384;

    k_impute<<<dim3(16), dim3(256), 0, stream>>>(
        xdyn, xmask, xdelta, gW, (const float*)d_in[5], ilW, (const float*)d_in[7],
        iWih, iWhh, (const float*)d_in[10], (const float*)d_in[11],
        o_xh, o_xc, xc_bf);

    k_lstm_t<128><<<dim3(16), dim3(256), 0, stream>>>(xc_bf, lWih[0], lWhh[0],
        (const float*)d_in[14], (const float*)d_in[15], lbuf0);
    k_lstm_t<256><<<dim3(16), dim3(256), 0, stream>>>(lbuf0, lWih[1], lWhh[1],
        (const float*)d_in[18], (const float*)d_in[19], lbuf1);
    k_lstm_t<256><<<dim3(16), dim3(256), 0, stream>>>(lbuf1, lWih[2], lWhh[2],
        (const float*)d_in[22], (const float*)d_in[23], lbuf0);

    k_feat<<<dim3((32768 * 320 + 255) / 256), dim3(256), 0, stream>>>(lbuf0, xstatic, feat);

    auto gemm = [&](const bf16* X, int K, const bf16* W, const float* bias, int N,
                    bf16* Ybf, float* Yf32, int mode) {
        k_gemm<<<dim3(2048), dim3(256), 0, stream>>>(X, K, W, bias, N, Ybf, Yf32, mode);
    };

    gemm(feat, 320, aux0, (const float*)d_in[25], 256, mp0, nullptr, 0);
    gemm(mp0,  256, aux1, (const float*)d_in[27], 256, mp1, nullptr, 0);
    gemm(mp1,  256, aux2, (const float*)d_in[29], 256, mp0, nullptr, 0);
    gemm(mp0,  256, auxO, (const float*)d_in[31], 16,  nullptr, aux_raw, 1);
    gemm(feat, 320, att0, (const float*)d_in[33], 256, mp0, nullptr, 0);
    gemm(mp0,  256, att1, (const float*)d_in[35], 256, mp1, nullptr, 0);
    gemm(mp1,  256, att2, (const float*)d_in[37], 256, mp0, nullptr, 0);
    gemm(mp0,  256, attO, attB,                    48,  nullptr, att_raw, 1);

    k_finalize<<<dim3(128), dim3(256), 0, stream>>>(aux_raw, att_raw, xstatic, xdyn,
                                                    o_aux, o_att, o_yp, o_exp);
}
